// CaptionGenerator_10144712753733
// MI455X (gfx1250) — matrix operand order repaired and checked by reading, not compiled, not run on hardware
//
#include <hip/hip_runtime.h>

// ---------------- types & WMMA helpers (CDNA5 / gfx1250, wave32) ----------------
typedef __attribute__((ext_vector_type(8)))  float  v8f;
typedef __attribute__((ext_vector_type(16))) __bf16 v16bf;
typedef __attribute__((ext_vector_type(8)))  __bf16 v8bf;

__device__ __forceinline__ __bf16 tobf(float x) {
    union { float f; unsigned u; } cv; cv.f = x;
    unsigned r = cv.u + 0x7FFFu + ((cv.u >> 16) & 1u);
    unsigned short s = (unsigned short)(r >> 16);
    union { unsigned short s; __bf16 b; } o; o.s = s;
    return o.b;
}

__device__ __forceinline__ v8f wmma_bf(v16bf a, v16bf b, v8f c) {
    return __builtin_amdgcn_wmma_f32_16x16x32_bf16(false, a, false, b, (short)0, c, false, false);
}

// A fragment, 16x32 bf16 (M x K), row-major source with leading dim lda.
// lane l: m = l%16 ; half = l/16 contributes +8 to k-base; lane holds k 0..7 and 16..23 (relative)
__device__ __forceinline__ v16bf load_frag_A(const __bf16* A, int lda, int m0, int kbase) {
    int lane = threadIdx.x & 31;
    const __bf16* row = A + (size_t)(m0 + (lane & 15)) * lda + kbase + ((lane >> 4) << 3);
    v8bf lo = *(const v8bf*)row;          // k 0..7
    v8bf hi = *(const v8bf*)(row + 16);   // k 16..23
    return __builtin_shufflevector(lo, hi, 0,1,2,3,4,5,6,7,8,9,10,11,12,13,14,15);
}

// B fragment, 32x16 bf16 (K x N). Column n of B == row n of BT (row-major, leading dim ldb).
// lane l: n = l%16 ; half = l/16 contributes +8 to k-base; lane holds k 0..7 and 16..23 (relative)
__device__ __forceinline__ v16bf load_frag_B(const __bf16* BT, int ldb, int n0, int kbase) {
    int lane = threadIdx.x & 31;
    const __bf16* row = BT + (size_t)(n0 + (lane & 15)) * ldb + kbase + ((lane >> 4) << 3);
    v8bf lo = *(const v8bf*)row;
    v8bf hi = *(const v8bf*)(row + 16);
    return __builtin_shufflevector(lo, hi, 0,1,2,3,4,5,6,7,8,9,10,11,12,13,14,15);
}

// ---------------- problem constants ----------------
#define BB 64
#define TT 20
#define TM 19          // Tmax
#define PP 36
#define VV 32000
#define EE 300
#define EP 320         // E padded to mult of 32
#define HH 512
#define DD 512
#define KIN 1324       // E + D + H
#define KINP 1344      // padded
#define NG 2048        // 4H

// output layout (floats)
#define PRED_OFF 0ll
#define ATT_OFF  (64ll*19*32000)
#define CAPS_OFF (ATT_OFF + 64ll*19*36)
#define DEC_OFF  (CAPS_OFF + 64ll*19)
#define SORT_OFF (DEC_OFF + 64ll)

// ---------------- workspace layout ----------------
constexpr size_t OFF_EMB_BF = 0;                               // V x 320 bf16
constexpr size_t OFF_WCAT   = OFF_EMB_BF + (size_t)VV*EP*2;    // 2048 x 1344 bf16
constexpr size_t OFF_C0     = OFF_WCAT   + (size_t)NG*KINP*2;  // B x V f32
constexpr size_t OFF_PROPS  = OFF_C0     + (size_t)BB*VV*4;    // B x P x D f32 (sorted)
constexpr size_t OFF_RH     = OFF_PROPS  + (size_t)BB*PP*DD*4; // B x P x H f32
constexpr size_t OFF_GATES  = OFF_RH     + (size_t)BB*PP*HH*4; // B x 2048 f32
constexpr size_t OFF_H      = OFF_GATES  + (size_t)BB*NG*4;
constexpr size_t OFF_CC     = OFF_H      + (size_t)BB*HH*4;
constexpr size_t OFF_FB     = OFF_CC     + (size_t)BB*HH*4;
constexpr size_t OFF_S      = OFF_FB     + (size_t)BB*DD*4;    // B x D
constexpr size_t OFF_XIN    = OFF_S      + (size_t)BB*DD*4;    // B x 1344 bf16
constexpr size_t OFF_UBF    = OFF_XIN    + (size_t)BB*KINP*2;  // B x 320 bf16
constexpr size_t OFF_SWRBF  = OFF_UBF    + (size_t)BB*EP*2;    // B x 320 bf16
constexpr size_t OFF_WLIN   = OFF_SWRBF  + (size_t)BB*EP*2;    // V f32
constexpr size_t OFF_COLSUM = OFF_WLIN   + (size_t)VV*4;       // 300 f32 (pad 1280)
constexpr size_t OFF_SUMWH  = OFF_COLSUM + 1280;               // H f32
constexpr size_t OFF_SUMWRP = OFF_SUMWH  + (size_t)HH*4;       // D f32
constexpr size_t OFF_GB     = OFF_SUMWRP + (size_t)DD*4;       // 2048 f32
constexpr size_t OFF_RLIN   = OFF_GB     + (size_t)NG*4;       // B x P f32
constexpr size_t OFF_WRSV   = OFF_RLIN   + (size_t)BB*PP*4;    // B x P f32
constexpr size_t OFF_BASE   = OFF_WRSV   + (size_t)BB*PP*4;    // B f32
constexpr size_t OFF_SDOTB  = OFF_BASE   + 256;                // B f32
constexpr size_t OFF_SWLIN  = OFF_SDOTB  + 256;                // 1 f32 (pad 64)
constexpr size_t OFF_SORT   = OFF_SWLIN  + 64;                 // B i32
constexpr size_t OFF_DEC    = OFF_SORT   + 256;                // B i32
constexpr size_t OFF_CAPS   = OFF_DEC    + 256;                // B x T i32

// ================= P0: stable descending sort + integer outputs =================
__global__ void sort_kernel(const int* lens, const int* captions, float* out,
                            int* sortind, int* dec_i, int* caps_s) {
    int i = threadIdx.x;              // 64 threads
    int li = lens[i];
    int rank = 0;
    for (int j = 0; j < BB; ++j) {
        int lj = lens[j];
        if (lj > li || (lj == li && j < i)) rank++;
    }
    sortind[rank] = i;
    dec_i[rank]   = li - 1;
    out[SORT_OFF + rank] = (float)i;
    out[DEC_OFF  + rank] = (float)(li - 1);
    for (int t = 0; t < TT; ++t) caps_s[rank*TT + t] = captions[i*TT + t];
    for (int t = 1; t < TT; ++t) out[CAPS_OFF + (long long)rank*TM + (t-1)] = (float)captions[i*TT + t];
}

// ================= P1: gather sorted rows, init state =================
__global__ void gather_kernel(const float* h0, const float* props, const int* sortind,
                              float* props_s, float* hbuf, float* cbuf, float* fbbuf) {
    long long g = (long long)blockIdx.x*blockDim.x + threadIdx.x;
    long long stride = (long long)gridDim.x*blockDim.x;
    const long long NP = (long long)BB*PP*DD;
    for (long long idx = g; idx < NP; idx += stride) {
        int b = (int)(idx / (PP*DD)); int rest = (int)(idx % (PP*DD));
        props_s[idx] = props[(long long)sortind[b]*PP*DD + rest];
    }
    for (long long idx = g; idx < BB*HH; idx += stride) {
        int b = (int)(idx >> 9); int d = (int)(idx & 511);
        float v = h0[sortind[b]*HH + d];
        hbuf[idx] = v; cbuf[idx] = v; fbbuf[idx] = 0.f;
    }
}

// ================= P2: emb_W -> bf16 padded =================
__global__ void embconv_kernel(const float* emb, __bf16* emb_bf) {
    long long g = (long long)blockIdx.x*blockDim.x + threadIdx.x;
    long long stride = (long long)gridDim.x*blockDim.x;
    const long long total = (long long)VV*EP;
    for (long long idx = g; idx < total; idx += stride) {
        int v = (int)(idx / EP), e = (int)(idx % EP);
        emb_bf[idx] = tobf(e < EE ? emb[(size_t)v*EE + e] : 0.f);
    }
}

// ================= P2b: w_lin[v] = emb_W[v]·w_W + w_b (wave per row) =================
__global__ void wlin_kernel(const float* emb, const float* w_W, const float* w_b, float* w_lin) {
    int wid = threadIdx.x >> 5, lane = threadIdx.x & 31;
    int v = blockIdx.x * 8 + wid;
    const float* row = emb + (size_t)v * EE;
    float s = 0.f;
    for (int e = lane; e < EE; e += 32) s += row[e] * w_W[e];
    for (int off = 16; off; off >>= 1) s += __shfl_down(s, off, 32);
    if (lane == 0) w_lin[v] = s + w_b[0];
}

// ================= P2c: colsum[e] = sum_v emb_W[v,e] (deterministic) =================
__global__ void colsum_kernel(const float* emb, float* colsum) {
    int e = blockIdx.x, tid = threadIdx.x;
    __shared__ float red[256];
    float s = 0.f;
    for (int v = tid; v < VV; v += 256) s += emb[(size_t)v*EE + e];
    red[tid] = s; __syncthreads();
    for (int k = 128; k; k >>= 1) { if (tid < k) red[tid] += red[tid+k]; __syncthreads(); }
    if (tid == 0) colsum[e] = red[0];
}

// ================= P2d: Wcat = [W_ih | W_hh] -> bf16 padded ; gate bias =================
__global__ void wcat_kernel(const float* W_ih, const float* W_hh, const float* b_ih,
                            const float* b_hh, __bf16* Wcat, float* gb) {
    long long g = (long long)blockIdx.x*blockDim.x + threadIdx.x;
    long long stride = (long long)gridDim.x*blockDim.x;
    const long long total = (long long)NG*KINP;
    for (long long idx = g; idx < total; idx += stride) {
        int n = (int)(idx / KINP), k = (int)(idx % KINP);
        float v = (k < 812) ? W_ih[(size_t)n*812 + k]
                            : (k < KIN ? W_hh[(size_t)n*HH + (k-812)] : 0.f);
        Wcat[idx] = tobf(v);
    }
    for (long long idx = g; idx < NG; idx += stride) gb[idx] = b_ih[idx] + b_hh[idx];
}

// ================= P3: S, sum_WH, sum_WRp, sum_wlin, Swr(bf16), Sdotb =================
__global__ void small_pre_kernel(const float* props_s, const float* colsum,
    const float* wh_W, const float* wh_b, const float* wr_W, const float* wr_b,
    const float* w_W, const float* w_b, float* S, float* sum_WH, float* sum_WRp,
    float* sum_wlin, __bf16* swr_bf, float* sdotb) {
    int tid = threadIdx.x;   // 512 threads
    for (int idx = tid; idx < BB*DD; idx += 512) {
        int b = idx >> 9, d = idx & 511;
        float s = 0.f;
        for (int p = 0; p < PP; ++p) s += props_s[((size_t)(b*PP + p))*DD + d];
        S[idx] = s;
    }
    {
        int h = tid;
        float s1 = 0.f, s2 = 0.f;
        for (int e = 0; e < EE; ++e) { float cs = colsum[e]; s1 += cs * wh_W[h*EE + e]; s2 += cs * wr_W[h*EE + e]; }
        sum_WH[h]  = s1 + (float)VV * wh_b[h];
        sum_WRp[h] = s2 + (float)VV * wr_b[h];
    }
    if (tid == 0) {
        float s = 0.f;
        for (int e = 0; e < EE; ++e) s += colsum[e] * w_W[e];
        sum_wlin[0] = s + (float)VV * w_b[0];
    }
    __syncthreads();
    for (int idx = tid; idx < BB*EP; idx += 512) {
        int b = idx / EP, e = idx % EP;
        float s = 0.f;
        if (e < EE) for (int d = 0; d < DD; ++d) s += S[b*DD + d] * wr_W[d*EE + e];
        swr_bf[idx] = tobf(e < EE ? s : 0.f);
    }
    for (int b = tid; b < BB; b += 512) {
        float s = 0.f;
        for (int d = 0; d < DD; ++d) s += S[b*DD + d] * wr_b[d];
        sdotb[b] = s;
    }
}

// ================= P4: RH, r_lin, wr_sumV (block per b, chunked p) =================
__global__ void rh_kernel(const float* props_s, const float* rh_W, const float* rh_b,
                          const float* r_W, const float* r_b, const float* sum_WRp,
                          float* RH, float* r_lin, float* wr_sumV) {
    int b = blockIdx.x, tid = threadIdx.x;
    __shared__ float pr[12*512];
    __shared__ float2 red2[256];
    for (int pc = 0; pc < PP; pc += 12) {
        for (int idx = tid; idx < 12*512; idx += 256) {
            int p = idx >> 9, d = idx & 511;
            pr[idx] = props_s[((size_t)(b*PP + pc + p))*DD + d];
        }
        __syncthreads();
        for (int h = tid; h < HH; h += 256) {
            float acc[12];
            #pragma unroll
            for (int p = 0; p < 12; ++p) acc[p] = 0.f;
            const float* wrow = rh_W + (size_t)h*DD;
            for (int d = 0; d < DD; ++d) {
                float w = wrow[d];
                #pragma unroll
                for (int p = 0; p < 12; ++p) acc[p] += w * pr[p*512 + d];
            }
            float bb = rh_b[h];
            for (int p = 0; p < 12; ++p) RH[((size_t)(b*PP + pc + p))*HH + h] = acc[p] + bb;
        }
        __syncthreads();
        for (int p = 0; p < 12; ++p) {
            float a = 0.f, c2 = 0.f;
            for (int d = tid; d < DD; d += 256) { a += pr[p*512+d]*r_W[d]; c2 += pr[p*512+d]*sum_WRp[d]; }
            red2[tid] = make_float2(a, c2); __syncthreads();
            for (int k = 128; k; k >>= 1) {
                if (tid < k) { red2[tid].x += red2[tid+k].x; red2[tid].y += red2[tid+k].y; }
                __syncthreads();
            }
            if (tid == 0) { r_lin[b*PP + pc + p] = red2[0].x + r_b[0]; wr_sumV[b*PP + pc + p] = red2[0].y; }
            __syncthreads();
        }
        __syncthreads();
    }
}

// ================= P6: initial xin (t=0): [emb(cap0) | fb=0 | h0s] -> bf16 =================
__global__ void initxin_kernel(const float* emb, const float* hbuf, const int* caps_s, __bf16* xin_bf) {
    int b = blockIdx.x, tid = threadIdx.x;
    int cap = caps_s[b*TT];
    const float* erow = emb + (size_t)cap*EE;
    for (int k = tid; k < KINP; k += 256) {
        float v = (k < EE) ? erow[k] : (k < 812 ? 0.f : (k < KIN ? hbuf[b*HH + (k-812)] : 0.f));
        xin_bf[b*KINP + k] = tobf(v);
    }
}

// ================= S1: gates = xin @ Wcat.T + gb  (WMMA bf16) =================
__global__ void gates_kernel(const __bf16* __restrict__ xin, const __bf16* __restrict__ Wc,
                             const float* gb, float* gates) {
    int wid = threadIdx.x >> 5, lane = threadIdx.x & 31;
    int m0 = (wid & 3) * 16;
    int n0 = blockIdx.x * 32 + (wid >> 2) * 16;
    v8f acc = {};
    for (int k = 0; k < KINP; k += 32) {
        v16bf af = load_frag_A(xin, KINP, m0, k);
        v16bf bf = load_frag_B(Wc,  KINP, n0, k);
        acc = wmma_bf(af, bf, acc);
    }
    int n = n0 + (lane & 15), half = lane >> 4;
    float g = gb[n];
    #pragma unroll
    for (int i = 0; i < 8; ++i)
        gates[(size_t)(m0 + i + 8*half)*NG + n] = acc[i] + g;
}

// ================= S2: LSTM nonlinearity + attention + stage next operands =================
__global__ void lstm_att_kernel(
    const float* gates, float* hbuf, float* cbuf, float* fbbuf,
    const float* RH, const float* r_lin, const float* wr_sumV,
    const float* sum_WH, const float* wh_b, const float* sum_wlin_p,
    const float* props_s, const float* wh_W, const float* emb,
    const int* caps_s, const int* dec_i,
    __bf16* u_bf, __bf16* xin_bf, float* base, float* att_out, int t) {
    int b = blockIdx.x, tid = threadIdx.x;
    bool mask = (t < dec_i[b]);
    __shared__ float hn[512], hu[512], fbu[512];
    __shared__ float rs[36], att_s[36];
    __shared__ float2 red2[256];

    for (int d = tid; d < HH; d += 256) {
        float gi = gates[(size_t)b*NG + d];
        float gf = gates[(size_t)b*NG + 512 + d];
        float gg = gates[(size_t)b*NG + 1024 + d];
        float go = gates[(size_t)b*NG + 1536 + d];
        float si = 1.f/(1.f + __expf(-gi));
        float sf = 1.f/(1.f + __expf(-gf));
        float tg = tanhf(gg);
        float so = 1.f/(1.f + __expf(-go));
        float c_old = cbuf[b*HH + d], h_old = hbuf[b*HH + d];
        float cn = sf*c_old + si*tg;
        float hv = so * tanhf(cn);
        hn[d] = hv;
        hu[d] = mask ? hv : h_old;
        if (mask) { cbuf[b*HH + d] = cn; hbuf[b*HH + d] = hv; }
    }
    __syncthreads();
    // r[p] = RH[b,p,:]·h_new + r_lin  (wave per p)
    int wid = tid >> 5, lane = tid & 31;
    for (int p = wid; p < PP; p += 8) {
        const float* rrow = RH + ((size_t)(b*PP + p))*HH;
        float s = 0.f;
        for (int hh = lane; hh < HH; hh += 32) s += rrow[hh]*hn[hh];
        for (int off = 16; off; off >>= 1) s += __shfl_down(s, off, 32);
        if (lane == 0) rs[p] = s + r_lin[b*PP + p];
    }
    // hsw = h_new·sum_WH ; hwb = h_new·wh_b   (block reduce)
    float p1 = 0.f, p2 = 0.f;
    for (int hh = tid; hh < HH; hh += 256) { p1 += hn[hh]*sum_WH[hh]; p2 += hn[hh]*wh_b[hh]; }
    red2[tid] = make_float2(p1, p2);
    __syncthreads();
    for (int k = 128; k; k >>= 1) {
        if (tid < k) { red2[tid].x += red2[tid+k].x; red2[tid].y += red2[tid+k].y; }
        __syncthreads();
    }
    if (tid == 0) {
        float hsw = red2[0].x + sum_wlin_p[0];
        float hwb = red2[0].y;
        float rsum = 0.f, mx = -1e30f;
        float lg[36];
        for (int p = 0; p < PP; ++p) {
            rsum += rs[p];
            float l = hsw + wr_sumV[b*PP + p] + rs[p];
            lg[p] = l; if (l > mx) mx = l;
        }
        float se = 0.f;
        for (int p = 0; p < PP; ++p) { float e = __expf(lg[p]-mx); att_s[p] = e; se += e; }
        float inv = 1.f/se;
        for (int p = 0; p < PP; ++p) att_s[p] *= inv;
        base[b] = rsum + hwb;
    }
    __syncthreads();
    float mf = mask ? 1.f : 0.f;
    if (tid < PP) att_out[((size_t)b*TM + t)*PP + tid] = att_s[tid]*mf;
    // fb_new = att @ props[b]
    for (int d = tid; d < DD; d += 256) {
        float s = 0.f;
        for (int p = 0; p < PP; ++p) s += att_s[p]*props_s[((size_t)(b*PP + p))*DD + d];
        float v = mask ? s : fbbuf[b*DD + d];
        fbu[d] = v;
        fbbuf[b*DD + d] = v;
    }
    // u = h_new @ wh_W  -> bf16 padded
    for (int e = tid; e < EP; e += 256) {
        float s = 0.f;
        if (e < EE) { for (int hh = 0; hh < HH; ++hh) s += hn[hh]*wh_W[hh*EE + e]; }
        u_bf[b*EP + e] = tobf(e < EE ? s : 0.f);
    }
    __syncthreads();
    // stage xin for next step: [emb(cap_{t+1}) | fb | h]
    if (t + 1 < TM) {
        int cap = caps_s[b*TT + t + 1];
        const float* erow = emb + (size_t)cap*EE;
        for (int k = tid; k < KIN; k += 256) {
            float v = (k < EE) ? erow[k] : (k < 812 ? fbu[k-EE] : hu[k-812]);
            xin_bf[b*KINP + k] = tobf(v);
        }
    }
}

// ================= S3 / P5: (64x320) @ emb_bf.T  (WMMA bf16) with epilogues =================
__global__ void gemm_embT_kernel(const __bf16* __restrict__ A, const __bf16* __restrict__ Bm,
                                 float* outp, long long ldo,
                                 const float* per_bv, const float* per_v, const float* per_b,
                                 const int* dec_i, int t) {
    int wid = threadIdx.x >> 5, lane = threadIdx.x & 31;
    int n0 = blockIdx.x * 128 + wid * 16;
    v8f acc0 = {}, acc1 = {}, acc2 = {}, acc3 = {};
    for (int k = 0; k < EP; k += 32) {
        v16bf bf = load_frag_B(Bm, EP, n0, k);
        v16bf a0 = load_frag_A(A, EP, 0,  k);
        v16bf a1 = load_frag_A(A, EP, 16, k);
        v16bf a2 = load_frag_A(A, EP, 32, k);
        v16bf a3 = load_frag_A(A, EP, 48, k);
        acc0 = wmma_bf(a0, bf, acc0);
        acc1 = wmma_bf(a1, bf, acc1);
        acc2 = wmma_bf(a2, bf, acc2);
        acc3 = wmma_bf(a3, bf, acc3);
    }
    int n = n0 + (lane & 15), half = lane >> 4;
    float pv = per_v ? per_v[n] : 0.f;
    v8f accs[4] = {acc0, acc1, acc2, acc3};
    #pragma unroll
    for (int mt = 0; mt < 4; ++mt) {
        #pragma unroll
        for (int i = 0; i < 8; ++i) {
            int b = mt*16 + i + 8*half;
            float val = accs[mt][i] + pv;
            if (per_b)  val += per_b[b];
            if (per_bv) val += per_bv[(size_t)b*VV + n];
            if (dec_i)  val = (t < dec_i[b]) ? val : 0.f;
            outp[(long long)b*ldo + n] = val;
        }
    }
}

// ================= host orchestration =================
extern "C" void kernel_launch(void* const* d_in, const int* in_sizes, int n_in,
                              void* d_out, int out_size, void* d_ws, size_t ws_size,
                              hipStream_t stream) {
    (void)in_sizes; (void)n_in; (void)out_size; (void)ws_size;
    const float* h0      = (const float*)d_in[0];
    const float* props   = (const float*)d_in[1];
    const int*   caps    = (const int*)d_in[2];
    const int*   lens    = (const int*)d_in[3];
    const float* emb     = (const float*)d_in[4];
    const float* W_ih    = (const float*)d_in[5];
    const float* W_hh    = (const float*)d_in[6];
    const float* b_ih    = (const float*)d_in[7];
    const float* b_hh    = (const float*)d_in[8];
    const float* wh_W    = (const float*)d_in[9];
    const float* wh_b    = (const float*)d_in[10];
    const float* wr_W    = (const float*)d_in[11];
    const float* wr_b    = (const float*)d_in[12];
    const float* rh_W    = (const float*)d_in[13];
    const float* rh_b    = (const float*)d_in[14];
    const float* w_W     = (const float*)d_in[15];
    const float* w_b     = (const float*)d_in[16];
    const float* r_W     = (const float*)d_in[17];
    const float* r_b     = (const float*)d_in[18];
    float* out = (float*)d_out;
    char*  ws  = (char*)d_ws;

    __bf16* emb_bf  = (__bf16*)(ws + OFF_EMB_BF);
    __bf16* wcat_bf = (__bf16*)(ws + OFF_WCAT);
    float*  C0      = (float*)(ws + OFF_C0);
    float*  props_s = (float*)(ws + OFF_PROPS);
    float*  RH      = (float*)(ws + OFF_RH);
    float*  gates   = (float*)(ws + OFF_GATES);
    float*  hbuf    = (float*)(ws + OFF_H);
    float*  cbuf    = (float*)(ws + OFF_CC);
    float*  fbbuf   = (float*)(ws + OFF_FB);
    float*  S       = (float*)(ws + OFF_S);
    __bf16* xin_bf  = (__bf16*)(ws + OFF_XIN);
    __bf16* u_bf    = (__bf16*)(ws + OFF_UBF);
    __bf16* swr_bf  = (__bf16*)(ws + OFF_SWRBF);
    float*  w_lin   = (float*)(ws + OFF_WLIN);
    float*  colsum  = (float*)(ws + OFF_COLSUM);
    float*  sum_WH  = (float*)(ws + OFF_SUMWH);
    float*  sum_WRp = (float*)(ws + OFF_SUMWRP);
    float*  gb      = (float*)(ws + OFF_GB);
    float*  r_lin   = (float*)(ws + OFF_RLIN);
    float*  wr_sumV = (float*)(ws + OFF_WRSV);
    float*  base    = (float*)(ws + OFF_BASE);
    float*  sdotb   = (float*)(ws + OFF_SDOTB);
    float*  swlin   = (float*)(ws + OFF_SWLIN);
    int*    sortind = (int*)(ws + OFF_SORT);
    int*    dec_i   = (int*)(ws + OFF_DEC);
    int*    caps_s  = (int*)(ws + OFF_CAPS);

    // precompute
    sort_kernel<<<1, 64, 0, stream>>>(lens, caps, out, sortind, dec_i, caps_s);
    gather_kernel<<<512, 256, 0, stream>>>(h0, props, sortind, props_s, hbuf, cbuf, fbbuf);
    embconv_kernel<<<2048, 256, 0, stream>>>(emb, emb_bf);
    wlin_kernel<<<4000, 256, 0, stream>>>(emb, w_W, w_b, w_lin);
    colsum_kernel<<<300, 256, 0, stream>>>(emb, colsum);
    wcat_kernel<<<512, 256, 0, stream>>>(W_ih, W_hh, b_ih, b_hh, wcat_bf, gb);
    small_pre_kernel<<<1, 512, 0, stream>>>(props_s, colsum, wh_W, wh_b, wr_W, wr_b,
                                            w_W, w_b, S, sum_WH, sum_WRp, swlin, swr_bf, sdotb);
    rh_kernel<<<64, 256, 0, stream>>>(props_s, rh_W, rh_b, r_W, r_b, sum_WRp, RH, r_lin, wr_sumV);
    // C0[b,v] = Swr@emb.T + w_lin[v] + Sdotb[b]   (== wr_sumP + w_lin)
    gemm_embT_kernel<<<250, 256, 0, stream>>>(swr_bf, emb_bf, C0, (long long)VV,
                                              nullptr, w_lin, sdotb, nullptr, 0);
    initxin_kernel<<<64, 256, 0, stream>>>(emb, hbuf, caps_s, xin_bf);

    // recurrent steps
    float* att_out = out + ATT_OFF;
    for (int t = 0; t < TM; ++t) {
        gates_kernel<<<64, 256, 0, stream>>>(xin_bf, wcat_bf, gb, gates);
        lstm_att_kernel<<<64, 256, 0, stream>>>(gates, hbuf, cbuf, fbbuf, RH, r_lin, wr_sumV,
                                                sum_WH, wh_b, swlin, props_s, wh_W, emb,
                                                caps_s, dec_i, u_bf, xin_bf, base, att_out, t);
        gemm_embT_kernel<<<250, 256, 0, stream>>>(u_bf, emb_bf,
                                                  out + (long long)t*VV, (long long)TM*VV,
                                                  C0, nullptr, base, dec_i, t);
    }
}
